// TinyGridMemorizer_68556267978868
// MI455X (gfx1250) — matrix operand order repaired and checked by reading, not compiled, not run on hardware
//
#include <hip/hip_runtime.h>
#include <hip/hip_bf16.h>

typedef __attribute__((ext_vector_type(16))) __bf16 v16bf;
typedef __attribute__((ext_vector_type(8)))  float  v8f;

#define DFEAT 4608
#define NMEM  50000
#define HID1  512
#define HID2  256
#define TOPK  50

// ---------- bf16 helpers (round-to-nearest-even) ----------
static __device__ __forceinline__ unsigned short f2bf(float f) {
    unsigned int u = __float_as_uint(f);
    u = u + 0x7FFFu + ((u >> 16) & 1u);
    return (unsigned short)(u >> 16);
}
static __device__ __forceinline__ unsigned int pack2bf(float a, float b) {
    unsigned int ua = __float_as_uint(a); ua = ua + 0x7FFFu + ((ua >> 16) & 1u);
    unsigned int ub = __float_as_uint(b); ub = ub + 0x7FFFu + ((ub >> 16) & 1u);
    return (ua >> 16) | (ub & 0xFFFF0000u);
}

// packed bf16: relu(a + b) on two bf16 lanes (CDNA5 VOP3P)
static __device__ __forceinline__ unsigned int pk_add_relu_bf16(unsigned int a, unsigned int b) {
    unsigned int s, r;
    asm("v_pk_add_bf16 %0, %1, %2" : "=v"(s) : "v"(a), "v"(b));
    asm("v_pk_max_num_bf16 %0, %1, 0" : "=v"(r) : "v"(s));
    return r;
}

// ---------- weight f32 -> bf16 conversion (strided sub-matrix) ----------
__global__ void cvt_bf16_kernel(const float* __restrict__ src, unsigned short* __restrict__ dst,
                                int total, int cols, int stride, int coff) {
    int i = blockIdx.x * blockDim.x + threadIdx.x;
    if (i >= total) return;
    int r = i / cols, c = i - r * cols;
    dst[i] = f2bf(src[(size_t)r * stride + coff + c]);
}

// ---------- tiny conv front-end ----------
__global__ void conv3x3_relu_kernel(const float* __restrict__ in, const float* __restrict__ w,
                                    const float* __restrict__ bias, float* __restrict__ out,
                                    int Cin, int Cout, int HW) {
    int b  = blockIdx.x / Cout;
    int co = blockIdx.x % Cout;
    int px = threadIdx.x;
    if (px >= HW * HW) return;
    int y = px / HW, x = px % HW;
    float acc = bias[co];
    for (int ci = 0; ci < Cin; ++ci) {
        const float* ip = in + ((size_t)b * Cin + ci) * HW * HW;
        const float* wp = w + ((size_t)co * Cin + ci) * 9;
        for (int ky = 0; ky < 3; ++ky) {
            int yy = y + ky - 1;
            if (yy < 0 || yy >= HW) continue;
            for (int kx = 0; kx < 3; ++kx) {
                int xx = x + kx - 1;
                if (xx < 0 || xx >= HW) continue;
                acc += ip[yy * HW + xx] * wp[ky * 3 + kx];
            }
        }
    }
    out[((size_t)b * Cout + co) * HW * HW + px] = fmaxf(acc, 0.f);
}

__global__ void pool5to3_kernel(const float* __restrict__ in, float* __restrict__ out) {
    int i = blockIdx.x * blockDim.x + threadIdx.x;   // over 4*256*9
    if (i >= 4 * 256 * 9) return;
    int px = i % 9, bc = i / 9;
    int oy = px / 3, ox = px % 3;
    const int st[3] = {0, 1, 3};
    const int ln[3] = {2, 3, 2};
    const float* p = in + (size_t)bc * 25;
    float s = 0.f;
    for (int y = st[oy]; y < st[oy] + ln[oy]; ++y)
        for (int x = st[ox]; x < st[ox] + ln[ox]; ++x)
            s += p[y * 5 + x];
    out[i] = s / (float)(ln[oy] * ln[ox]);
}

// ---------- small FC: out[b][o] = act(in[b]·w[o,woff:woff+K] + bias[o]) for b=0..3 ----------
__global__ void fc4_kernel(const float* __restrict__ in, int K,
                           const float* __restrict__ w, int wstride, int woff,
                           const float* __restrict__ bias, float* __restrict__ out,
                           int O, int doRelu) {
    int o = blockIdx.x;
    int tid = threadIdx.x;
    const float* wr = w + (size_t)o * wstride + woff;
    float a0 = 0, a1 = 0, a2 = 0, a3 = 0;
    for (int k = tid; k < K; k += 128) {
        float wv = wr[k];
        a0 += in[k] * wv;
        a1 += in[K + k] * wv;
        a2 += in[2 * K + k] * wv;
        a3 += in[3 * K + k] * wv;
    }
    __shared__ float sm[4][128];
    sm[0][tid] = a0; sm[1][tid] = a1; sm[2][tid] = a2; sm[3][tid] = a3;
    __syncthreads();
    for (int off = 64; off > 0; off >>= 1) {
        if (tid < off)
            for (int b = 0; b < 4; ++b) sm[b][tid] += sm[b][tid + off];
        __syncthreads();
    }
    if (tid < 4) {
        float v = sm[tid][0] + bias[o];
        if (doRelu) v = fmaxf(v, 0.f);
        out[(size_t)tid * O + o] = v;
    }
}

// ---------- kernel 1: kp[n][o] = mem[n]·w1k[o]  (bf16 WMMA, f32 acc) ----------
// grid.x = ceil(NMEM/64), block = 256 (8 waves); wave w covers o in [w*64, w*64+64),
// each wave processes 4 row-tiles (64 rows/block) so each B fragment feeds 4 WMMAs.
__global__ void __launch_bounds__(256) kp_gemm_kernel(const float* __restrict__ mem,
                                                      const unsigned short* __restrict__ w1k,
                                                      unsigned short* __restrict__ kp) {
    const int wave = threadIdx.x >> 5;
    const int lane = threadIdx.x & 31;
    const int laneLo = lane & 15;
    const int laneHi = lane >> 4;
    const int n0 = blockIdx.x * 64;
    const int o0 = wave * 64;

    const float* arow[4];
#pragma unroll
    for (int rt = 0; rt < 4; ++rt) {
        int r = n0 + rt * 16 + laneLo;
        if (r > NMEM - 1) r = NMEM - 1;      // clamp; stores are guarded below
        arow[rt] = mem + (size_t)r * DFEAT;
    }

    v8f acc[4][4] = {};   // [row-tile][col-tile]
    for (int k0 = 0; k0 < DFEAT; k0 += 32) {
        const int ka = k0 + laneHi * 8;
        union { v16bf v; unsigned int u[8]; } A[4];
#pragma unroll
        for (int rt = 0; rt < 4; ++rt) {
            const float* p = arow[rt] + ka;
            __builtin_prefetch(p + 256, 0, 3);   // ~1KB ahead -> global_prefetch
            float4 f0 = *(const float4*)(p);
            float4 f1 = *(const float4*)(p + 4);
            float4 f2 = *(const float4*)(p + 16);
            float4 f3 = *(const float4*)(p + 20);
            A[rt].u[0] = pack2bf(f0.x, f0.y); A[rt].u[1] = pack2bf(f0.z, f0.w);
            A[rt].u[2] = pack2bf(f1.x, f1.y); A[rt].u[3] = pack2bf(f1.z, f1.w);
            A[rt].u[4] = pack2bf(f2.x, f2.y); A[rt].u[5] = pack2bf(f2.z, f2.w);
            A[rt].u[6] = pack2bf(f3.x, f3.y); A[rt].u[7] = pack2bf(f3.z, f3.w);
        }
#pragma unroll
        for (int t = 0; t < 4; ++t) {
            const unsigned short* bp =
                w1k + (size_t)(o0 + t * 16 + laneLo) * DFEAT + k0 + laneHi * 8;
            union { v16bf v; uint4 q[2]; } B;
            B.q[0] = *(const uint4*)bp;
            B.q[1] = *(const uint4*)(bp + 16);
#pragma unroll
            for (int rt = 0; rt < 4; ++rt)
                acc[rt][t] = __builtin_amdgcn_wmma_f32_16x16x32_bf16(
                    false, A[rt].v, false, B.v, (short)0, acc[rt][t], false, false);
        }
    }
#pragma unroll
    for (int rt = 0; rt < 4; ++rt)
#pragma unroll
        for (int t = 0; t < 4; ++t)
#pragma unroll
            for (int r = 0; r < 8; ++r) {
                int row = n0 + rt * 16 + r + laneHi * 8;
                int col = o0 + t * 16 + laneLo;
                if (row < NMEM)
                    kp[(size_t)row * HID1 + col] = f2bf(acc[rt][t][r]);
            }
}

// ---------- kernel 2: fused h1=relu(qp+kp); h2=relu(W2·h1+b2); sim=h2·w3+b3 ----------
// grid.x = NMEM/16, block = 128 (4 waves); wave w covers o in [w*64, w*64+64) (256 total).
// qp pre-converted to bf16; relu(qp+kp) done with packed bf16 VALU (v_pk_add/v_pk_max_num).
__global__ void __launch_bounds__(128) sim_kernel(const unsigned short* __restrict__ kp,
                                                  const unsigned short* __restrict__ qp_bf,
                                                  const unsigned short* __restrict__ w2,
                                                  const float* __restrict__ b2,
                                                  const float* __restrict__ w3,
                                                  const float* __restrict__ b3,
                                                  float* __restrict__ sim) {
    __shared__ unsigned int qpl[4 * HID1 / 2];      // bf16 pairs: [b][256]
    __shared__ float part[4][4][16][16];            // [wave][batch][row][laneLo]
    {
        const unsigned int* qsrc = (const unsigned int*)qp_bf;
        for (int i = threadIdx.x; i < 4 * HID1 / 2; i += 128) qpl[i] = qsrc[i];
    }
    __syncthreads();

    const int wave = threadIdx.x >> 5;
    const int lane = threadIdx.x & 31;
    const int laneLo = lane & 15;
    const int laneHi = lane >> 4;
    const int n0 = blockIdx.x * 16;
    const int o0 = wave * 64;

    v8f acc[4][4] = {};   // [batch][tile]
    const unsigned short* krow = kp + (size_t)(n0 + laneLo) * HID1;

    for (int k0 = 0; k0 < HID1; k0 += 32) {
        const int ka = k0 + laneHi * 8;        // element index; pair index = ka/2
        uint4 c0 = *(const uint4*)(krow + ka);        // kp pairs u[0..3]
        uint4 c1 = *(const uint4*)(krow + ka + 16);   // kp pairs u[4..7]
        unsigned int kpu[8] = {c0.x, c0.y, c0.z, c0.w, c1.x, c1.y, c1.z, c1.w};

        union { v16bf v; unsigned int u[8]; } A[4];
#pragma unroll
        for (int b = 0; b < 4; ++b) {
            const unsigned int* q = qpl + b * (HID1 / 2) + (ka >> 1);
            uint4 q0 = *(const uint4*)(q);        // qp pairs for u[0..3]
            uint4 q1 = *(const uint4*)(q + 8);    // qp pairs for u[4..7]
            A[b].u[0] = pk_add_relu_bf16(kpu[0], q0.x);
            A[b].u[1] = pk_add_relu_bf16(kpu[1], q0.y);
            A[b].u[2] = pk_add_relu_bf16(kpu[2], q0.z);
            A[b].u[3] = pk_add_relu_bf16(kpu[3], q0.w);
            A[b].u[4] = pk_add_relu_bf16(kpu[4], q1.x);
            A[b].u[5] = pk_add_relu_bf16(kpu[5], q1.y);
            A[b].u[6] = pk_add_relu_bf16(kpu[6], q1.z);
            A[b].u[7] = pk_add_relu_bf16(kpu[7], q1.w);
        }
#pragma unroll
        for (int t = 0; t < 4; ++t) {
            const unsigned short* bp =
                w2 + (size_t)(o0 + t * 16 + laneLo) * HID1 + k0 + laneHi * 8;
            union { v16bf v; uint4 q[2]; } B;
            B.q[0] = *(const uint4*)bp;
            B.q[1] = *(const uint4*)(bp + 16);
#pragma unroll
            for (int b = 0; b < 4; ++b)
                acc[b][t] = __builtin_amdgcn_wmma_f32_16x16x32_bf16(
                    false, A[b].v, false, B.v, (short)0, acc[b][t], false, false);
        }
    }

    float biasv[4], w3v[4];
#pragma unroll
    for (int t = 0; t < 4; ++t) {
        int col = o0 + t * 16 + laneLo;
        biasv[t] = b2[col];
        w3v[t]   = w3[col];
    }
#pragma unroll
    for (int b = 0; b < 4; ++b)
#pragma unroll
        for (int r = 0; r < 8; ++r) {
            float s = 0.f;
#pragma unroll
            for (int t = 0; t < 4; ++t) {
                float v = fmaxf(acc[b][t][r] + biasv[t], 0.f);
                s += v * w3v[t];
            }
            part[wave][b][r + laneHi * 8][laneLo] = s;
        }
    __syncthreads();
    if (threadIdx.x < 64) {
        int b = threadIdx.x >> 4, row = threadIdx.x & 15;
        float s = 0.f;
        for (int w = 0; w < 4; ++w)
            for (int l = 0; l < 16; ++l) s += part[w][b][row][l];
        sim[(size_t)b * NMEM + n0 + row] = s + b3[0];
    }
}

// ---------- kernel 3: top-50 + softmax + one-hot gather + blend ----------
__global__ void topk_combine_kernel(float* __restrict__ sim, const int* __restrict__ memout,
                                    const float* __restrict__ decoded, float* __restrict__ out) {
    int b = blockIdx.x;
    int tid = threadIdx.x;
    float* s = sim + (size_t)b * NMEM;
    __shared__ float rv[256];
    __shared__ int ri[256];
    __shared__ float topv[TOPK];
    __shared__ int topi[TOPK];
    __shared__ float wts[TOPK];
    __shared__ float wsum;

    for (int it = 0; it < TOPK; ++it) {
        float best = -3.0e38f;
        int bi = 0;
        for (int i = tid; i < NMEM; i += 256) {
            float v = s[i];
            if (v > best) { best = v; bi = i; }
        }
        rv[tid] = best; ri[tid] = bi;
        __syncthreads();
        for (int off = 128; off > 0; off >>= 1) {
            if (tid < off) {
                if (rv[tid + off] > rv[tid] ||
                    (rv[tid + off] == rv[tid] && ri[tid + off] < ri[tid])) {
                    rv[tid] = rv[tid + off];
                    ri[tid] = ri[tid + off];
                }
            }
            __syncthreads();
        }
        if (tid == 0) {
            topv[it] = rv[0];
            topi[it] = ri[0];
            s[ri[0]] = -3.4e38f;   // regenerated by sim_kernel each launch
        }
        __syncthreads();
    }
    if (tid == 0) {
        float m = topv[0];   // iterative max -> descending, topv[0] is max
        float sum = 0.f;
        for (int i = 0; i < TOPK; ++i) {
            float e = expf((topv[i] - m) * 10.0f);
            wts[i] = e;
            sum += e;
        }
        wsum = sum;
    }
    __syncthreads();
    if (tid < 250) {
        int c = tid / 25, px = tid % 25;
        float acc = 0.f;
        for (int i = 0; i < TOPK; ++i) {
            int cls = memout[(size_t)topi[i] * 25 + px];
            if (cls == c) acc += wts[i];
        }
        acc /= wsum;
        out[(size_t)b * 250 + tid] = 0.8f * acc + 0.2f * decoded[(size_t)b * 250 + tid];
    }
}

// ---------- launch ----------
extern "C" void kernel_launch(void* const* d_in, const int* in_sizes, int n_in,
                              void* d_out, int out_size, void* d_ws, size_t ws_size,
                              hipStream_t stream) {
    (void)in_sizes; (void)n_in; (void)out_size; (void)ws_size;
    const float* x       = (const float*)d_in[0];
    const float* mem_in  = (const float*)d_in[1];
    const int*   mem_out = (const int*)d_in[2];
    const float* conv1_w = (const float*)d_in[3];
    const float* conv1_b = (const float*)d_in[4];
    const float* conv2_w = (const float*)d_in[5];
    const float* conv2_b = (const float*)d_in[6];
    const float* conv3_w = (const float*)d_in[7];
    const float* conv3_b = (const float*)d_in[8];
    const float* conv4_w = (const float*)d_in[9];
    const float* conv4_b = (const float*)d_in[10];
    const float* sim_w1  = (const float*)d_in[11];
    const float* sim_b1  = (const float*)d_in[12];
    const float* sim_w2  = (const float*)d_in[13];
    const float* sim_b2  = (const float*)d_in[14];
    const float* sim_w3  = (const float*)d_in[15];
    const float* sim_b3  = (const float*)d_in[16];
    const float* dec_w1  = (const float*)d_in[17];
    const float* dec_b1  = (const float*)d_in[18];
    const float* dec_w2  = (const float*)d_in[19];
    const float* dec_b2  = (const float*)d_in[20];
    float* out = (float*)d_out;

    char* ws = (char*)d_ws;
    size_t off = 0;
    auto alloc = [&](size_t bytes) -> char* {
        char* p = ws + off;
        off += (bytes + 255) & ~(size_t)255;
        return p;
    };
    unsigned short* kp_bf  = (unsigned short*)alloc((size_t)NMEM * HID1 * 2);    // 51.2 MB
    unsigned short* w1k_bf = (unsigned short*)alloc((size_t)HID1 * DFEAT * 2);   // 4.7 MB
    unsigned short* w2_bf  = (unsigned short*)alloc((size_t)HID2 * HID1 * 2);    // 256 KB
    unsigned short* qp_bf  = (unsigned short*)alloc((size_t)4 * HID1 * 2);
    float* sim = (float*)alloc((size_t)4 * NMEM * 4);                            // 800 KB
    float* h1b = (float*)alloc((size_t)4 * 64 * 25 * 4);
    float* h2b = (float*)alloc((size_t)4 * 128 * 25 * 4);
    float* h3b = (float*)alloc((size_t)4 * 256 * 25 * 4);
    float* hpb = (float*)alloc((size_t)4 * 256 * 9 * 4);
    float* enc = (float*)alloc((size_t)4 * DFEAT * 4);
    float* qp  = (float*)alloc((size_t)4 * HID1 * 4);
    float* d1  = (float*)alloc((size_t)4 * HID1 * 4);
    float* dec = (float*)alloc((size_t)4 * 250 * 4);

    // weight conversion to bf16 (w1k = sim_w1[:, D:2D], w2 = sim_w2)
    cvt_bf16_kernel<<<(HID1 * DFEAT + 255) / 256, 256, 0, stream>>>(
        sim_w1, w1k_bf, HID1 * DFEAT, DFEAT, 2 * DFEAT, DFEAT);
    cvt_bf16_kernel<<<(HID2 * HID1 + 255) / 256, 256, 0, stream>>>(
        sim_w2, w2_bf, HID2 * HID1, HID1, HID1, 0);

    // conv front-end
    conv3x3_relu_kernel<<<4 * 64, 32, 0, stream>>>(x, conv1_w, conv1_b, h1b, 10, 64, 5);
    conv3x3_relu_kernel<<<4 * 128, 32, 0, stream>>>(h1b, conv2_w, conv2_b, h2b, 64, 128, 5);
    conv3x3_relu_kernel<<<4 * 256, 32, 0, stream>>>(h2b, conv3_w, conv3_b, h3b, 128, 256, 5);
    pool5to3_kernel<<<(4 * 256 * 9 + 255) / 256, 256, 0, stream>>>(h3b, hpb);
    conv3x3_relu_kernel<<<4 * 512, 32, 0, stream>>>(hpb, conv4_w, conv4_b, enc, 256, 512, 3);

    // small FCs: qp = enc·w1q^T + b1 ; decoder
    fc4_kernel<<<HID1, 128, 0, stream>>>(enc, DFEAT, sim_w1, 2 * DFEAT, 0, sim_b1, qp, HID1, 0);
    fc4_kernel<<<HID1, 128, 0, stream>>>(enc, DFEAT, dec_w1, DFEAT, 0, dec_b1, d1, HID1, 1);
    fc4_kernel<<<250, 128, 0, stream>>>(d1, HID1, dec_w2, HID1, 0, dec_b2, dec, 250, 0);
    // qp -> bf16 for packed-bf16 fused add in sim_kernel
    cvt_bf16_kernel<<<(4 * HID1 + 255) / 256, 256, 0, stream>>>(qp, qp_bf, 4 * HID1, HID1, HID1, 0);

    // big WMMA GEMMs
    kp_gemm_kernel<<<(NMEM + 63) / 64, 256, 0, stream>>>(mem_in, w1k_bf, kp_bf);
    sim_kernel<<<NMEM / 16, 128, 0, stream>>>(kp_bf, qp_bf, w2_bf, sim_b2, sim_w3, sim_b3, sim);

    // top-k + softmax + blend
    topk_combine_kernel<<<4, 256, 0, stream>>>(sim, mem_out, dec, out);
}
